// GRUCELLTaskSpecific_76991583748616
// MI455X (gfx1250) — matrix operand order repaired and checked by reading, not compiled, not run on hardware
//
#include <hip/hip_runtime.h>

// ---------------------------------------------------------------------------
// Shapes
// ---------------------------------------------------------------------------
#define BATCH 262144
#define DIN   250
#define HID   100
#define KX    256          // DIN padded to 8 k-tiles of 32
#define KH    128          // HID padded to 4 k-tiles of 32
#define NT    7            // HID padded to 112 -> 7 n-tiles of 16

#define WX_ELEMS (6*8*NT*512)      // 6 x-side gate blocks, 8 ktiles, 7 ntiles, 512 bf16/frag
#define WH_ELEMS (3*4*NT*512)      // 3 gate blocks, 4 ktiles
#define TOT_PACK (WX_ELEMS + 3*WH_ELEMS)

typedef __attribute__((ext_vector_type(16))) __bf16        v16bf;
typedef __attribute__((ext_vector_type(4)))  __bf16        bf16x4;
typedef __attribute__((ext_vector_type(8)))  float         v8f;
typedef __attribute__((ext_vector_type(4)))  float         f32x4;
typedef __attribute__((ext_vector_type(4)))  unsigned int  u32x4;
typedef f32x4 __attribute__((aligned(4)))                  f32x4_u;  // rows only 4/8B aligned

union BFrag { v16bf v; u32x4 q[2]; };

__device__ __forceinline__ unsigned short f2bf(float f) {
  union { __bf16 b; unsigned short s; } u;
  u.b = (__bf16)f;                               // native RNE f32->bf16 cvt
  return u.s;
}
__device__ __forceinline__ float sigm(float x) { return 1.f / (1.f + __expf(-x)); }

// XOR-swizzle 16-byte chunks within an LDS row to break bank conflicts on
// ds_load_b128 A-fragment reads (row strides are multiples of 64 banks).
// Operates on 8-element chunks, so 4-element groups at k%4==0 stay contiguous.
__device__ __forceinline__ int swz(int m, int k) {
  return ((((k >> 3) ^ (m & 7)) << 3) | (k & 7));
}

__device__ __forceinline__ v8f wmma_bf16(v16bf a, v16bf b, v8f c) {
  // v_wmma_f32_16x16x32_bf16: (neg_a, A, neg_b, B, c_mod, C, reuse_a, reuse_b)
  return __builtin_amdgcn_wmma_f32_16x16x32_bf16(false, a, false, b, (short)0, c,
                                                 false, false);
}

// A-fragment (16x32 bf16): lane m=l%16; lanes 0-15 hold K {0..15},
// lanes 16-31 hold {16..31} (2 bf16 per VGPR, K ascending).
__device__ __forceinline__ v16bf load_a(const unsigned short* tile, int stride,
                                        int kt, int lane) {
  const int m = lane & 15, h = lane >> 4;
  const int k0 = kt * 32 + h * 16;
  const unsigned short* row = tile + m * stride;
  BFrag u;
  u.q[0] = *(const u32x4*)(row + swz(m, k0));
  u.q[1] = *(const u32x4*)(row + swz(m, k0 + 8));
  return u.v;
}

// B-fragment (32x16 bf16) packed lane-contiguous: lane l holds col n=l%16,
// 16 consecutive K values starting at (l/16)*16, stored as 32 contiguous bytes.
__device__ __forceinline__ v16bf load_b(const unsigned short* region, int frag,
                                        int lane) {
  const u32x4* p = (const u32x4*)(region + (frag << 9));
  BFrag u;
  u.q[0] = p[lane * 2];
  u.q[1] = p[lane * 2 + 1];
  return u.v;
}

// Stage a 16xHID f32 tile as 16xKH bf16 into swizzled LDS, 4-wide.
// k%4==0 and HID%4==0, so each 4-group is fully in-range (k<HID) or all-pad.
__device__ __forceinline__ void stage_h(unsigned short* dst, const float* __restrict__ src,
                                        long r0, int lane) {
  for (int idx4 = lane; idx4 < 16 * (KH / 4); idx4 += 32) {
    int m = idx4 >> 5;
    int k = (idx4 & 31) << 2;
    f32x4 v = {0.f, 0.f, 0.f, 0.f};
    if (k < HID) v = *(const f32x4_u*)(src + (r0 + m) * HID + k);
    *(bf16x4*)(dst + m * KH + swz(m, k)) = __builtin_convertvector(v, bf16x4);
  }
}

// ---------------------------------------------------------------------------
// Kernel 0: pack all weights (fp32 -> bf16) into WMMA B-fragment layout in d_ws
// Layout: [Wx: 6 blocks(ir,iz,in,Wz,Wr,Whn) x 8kt x 7nt]
//         [Wh: 3 blocks(r,z,n) x 4kt x 7nt] [U: z,r,hn] [Us: z,r,hn]
// ---------------------------------------------------------------------------
__global__ void pack_weights(const float* __restrict__ Wi, const float* __restrict__ Wh,
                             const float* __restrict__ W_z, const float* __restrict__ W_r,
                             const float* __restrict__ W_hn, const float* __restrict__ U_z,
                             const float* __restrict__ U_r, const float* __restrict__ U_hn,
                             const float* __restrict__ Us_z, const float* __restrict__ Us_r,
                             const float* __restrict__ Us_hn, unsigned short* __restrict__ wp) {
  int i = blockIdx.x * blockDim.x + threadIdx.x;
  if (i >= TOT_PACK) return;
  float v = 0.f;
  if (i < WX_ELEMS) {
    int b  = i / (8 * NT * 512);
    int r  = i % (8 * NT * 512);
    int kt = r / (NT * 512);  r %= NT * 512;
    int j  = r / 512;
    int f  = r % 512;
    int lane = f >> 4, e = f & 15;
    int n = j * 16 + (lane & 15);
    int k = kt * 32 + (lane >> 4) * 16 + e;
    if (n < HID && k < DIN) {
      if (b < 3) v = Wi[(b * HID + n) * DIN + k];
      else {
        const float* W = (b == 3) ? W_z : (b == 4) ? W_r : W_hn;
        v = W[n * DIN + k];
      }
    }
  } else {
    int t = i - WX_ELEMS;
    int region = t / WH_ELEMS;            // 0=Wh, 1=U, 2=Us
    int r = t % WH_ELEMS;
    int b  = r / (4 * NT * 512);  r %= 4 * NT * 512;
    int kt = r / (NT * 512);      r %= NT * 512;
    int j  = r / 512;
    int f  = r % 512;
    int lane = f >> 4, e = f & 15;
    int n = j * 16 + (lane & 15);
    int k = kt * 32 + (lane >> 4) * 16 + e;
    if (n < HID && k < HID) {
      if (region == 0)      v = Wh[(b * HID + n) * HID + k];
      else if (region == 1) { const float* U = (b == 0) ? U_z  : (b == 1) ? U_r  : U_hn;  v = U[n * HID + k]; }
      else                  { const float* U = (b == 0) ? Us_z : (b == 1) ? Us_r : Us_hn; v = U[n * HID + k]; }
    }
  }
  wp[i] = f2bf(v);
}

// ---------------------------------------------------------------------------
// Kernel 1: fused GRU stack. 4 wave32 per block; each wave owns 16 batch rows.
// ---------------------------------------------------------------------------
__global__ __launch_bounds__(128) void gru_fused(
    const float* __restrict__ x, const float* __restrict__ h_prev,
    const float* __restrict__ hs_prev, const float* __restrict__ bi,
    const float* __restrict__ bh, const unsigned short* __restrict__ wp,
    float* __restrict__ out) {
  __shared__ __align__(16) unsigned short sX[4][16 * KX];   // x tile, bf16
  __shared__ __align__(16) unsigned short sH[4][16 * KH];   // hs_prev then h_prev
  __shared__ __align__(16) unsigned short sN[4][16 * KH];   // h_shared_new, bf16

  const int tid = threadIdx.x, wave = tid >> 5, lane = tid & 31;
  const long r0 = (long)blockIdx.x * 64 + wave * 16;
  const int cm = lane & 15, ch = lane >> 4;

  // ---- stage x tile: f32x4 global loads -> packed bf16x4 -> ds_store_b64 ----
  for (int idx4 = lane; idx4 < 16 * (KX / 4); idx4 += 32) {
    int m = idx4 >> 6;
    int k = (idx4 & 63) << 2;
    f32x4 v;
    if (k + 3 < DIN) {
      v = *(const f32x4_u*)(x + (r0 + m) * DIN + k);
    } else {
#pragma unroll
      for (int t = 0; t < 4; ++t)
        v[t] = (k + t < DIN) ? x[(r0 + m) * DIN + k + t] : 0.f;
    }
    *(bf16x4*)(&sX[wave][m * KX + swz(m, k)]) = __builtin_convertvector(v, bf16x4);
  }
  // ---- stage h_shared_prev; zero h_shared_new buffer ----
  stage_h(sH[wave], hs_prev, r0, lane);
  for (int idx4 = lane; idx4 < 16 * (KH / 4); idx4 += 32) {
    bf16x4 z = {};
    *(bf16x4*)(&sN[wave][idx4 << 2]) = z;
  }
  __syncthreads();

  const unsigned short* WXp = wp;
  const unsigned short* WHp = wp + WX_ELEMS;
  const unsigned short* Up  = WHp + WH_ELEMS;
  const unsigned short* USp = Up + WH_ELEMS;

  // ================= Stage 1: shared GRU cell -> h_shared_new ==============
  for (int j = 0; j < NT; ++j) {
    v8f ar = {}, az = {}, an_ = {}, hr = {}, hz = {}, hn = {};
#pragma unroll
    for (int kt = 0; kt < 8; ++kt) {
      v16bf a = load_a(sX[wave], KX, kt, lane);
      ar  = wmma_bf16(a, load_b(WXp, (0 * 8 + kt) * NT + j, lane), ar);
      az  = wmma_bf16(a, load_b(WXp, (1 * 8 + kt) * NT + j, lane), az);
      an_ = wmma_bf16(a, load_b(WXp, (2 * 8 + kt) * NT + j, lane), an_);
    }
#pragma unroll
    for (int kt = 0; kt < 4; ++kt) {
      v16bf a = load_a(sH[wave], KH, kt, lane);
      hr = wmma_bf16(a, load_b(WHp, (0 * 4 + kt) * NT + j, lane), hr);
      hz = wmma_bf16(a, load_b(WHp, (1 * 4 + kt) * NT + j, lane), hz);
      hn = wmma_bf16(a, load_b(WHp, (2 * 4 + kt) * NT + j, lane), hn);
    }
    const int n = j * 16 + cm;
    const bool nv = n < HID;
    float bir = nv ? bi[n] : 0.f, biz = nv ? bi[HID + n] : 0.f, bin = nv ? bi[2 * HID + n] : 0.f;
    float bhr = nv ? bh[n] : 0.f, bhz = nv ? bh[HID + n] : 0.f, bhn = nv ? bh[2 * HID + n] : 0.f;
#pragma unroll
    for (int e = 0; e < 8; ++e) {
      const int M = ch * 8 + e;                 // C layout: VGPR e, half ch
      float hsp = nv ? hs_prev[(r0 + M) * HID + n] : 0.f;
      float rr = sigm(ar[e] + bir + hr[e] + bhr);
      float zz = sigm(az[e] + biz + hz[e] + bhz);
      float ng = tanhf(an_[e] + bin + rr * (hn[e] + bhn));
      float v = nv ? ((1.f - zz) * ng + zz * hsp) : 0.f;
      sN[wave][M * KH + swz(M, n)] = f2bf(v);
    }
  }
  __syncthreads();
  // ---- restage sH with h_prev ----
  stage_h(sH[wave], h_prev, r0, lane);
  __syncthreads();

  // ================= Stage 2: task-specific gates -> h_new =================
  for (int j = 0; j < NT; ++j) {
    v8f xz = {}, xr = {}, xn = {}, uz = {}, ur = {}, un = {}, sz = {}, sr = {}, sn = {};
#pragma unroll
    for (int kt = 0; kt < 8; ++kt) {
      v16bf a = load_a(sX[wave], KX, kt, lane);
      xz = wmma_bf16(a, load_b(WXp, (3 * 8 + kt) * NT + j, lane), xz);
      xr = wmma_bf16(a, load_b(WXp, (4 * 8 + kt) * NT + j, lane), xr);
      xn = wmma_bf16(a, load_b(WXp, (5 * 8 + kt) * NT + j, lane), xn);
    }
#pragma unroll
    for (int kt = 0; kt < 4; ++kt) {
      v16bf a = load_a(sH[wave], KH, kt, lane);
      uz = wmma_bf16(a, load_b(Up, (0 * 4 + kt) * NT + j, lane), uz);
      ur = wmma_bf16(a, load_b(Up, (1 * 4 + kt) * NT + j, lane), ur);
      un = wmma_bf16(a, load_b(Up, (2 * 4 + kt) * NT + j, lane), un);
    }
#pragma unroll
    for (int kt = 0; kt < 4; ++kt) {
      v16bf a = load_a(sN[wave], KH, kt, lane);
      sz = wmma_bf16(a, load_b(USp, (0 * 4 + kt) * NT + j, lane), sz);
      sr = wmma_bf16(a, load_b(USp, (1 * 4 + kt) * NT + j, lane), sr);
      sn = wmma_bf16(a, load_b(USp, (2 * 4 + kt) * NT + j, lane), sn);
    }
    const int n = j * 16 + cm;
    if (n < HID) {
#pragma unroll
      for (int e = 0; e < 8; ++e) {
        const int M = ch * 8 + e;
        const long row = r0 + M;
        float hp = h_prev[row * HID + n];
        float zz = sigm(xz[e] + uz[e] + sz[e]);
        float rr = sigm(xr[e] + ur[e] + sr[e]);
        float nm = tanhf(xn[e] + rr * un[e] + sn[e]);
        out[row * HID + n] = (1.f - zz) * nm + zz * hp;
      }
    }
  }
}

// ---------------------------------------------------------------------------
extern "C" void kernel_launch(void* const* d_in, const int* in_sizes, int n_in,
                              void* d_out, int out_size, void* d_ws, size_t ws_size,
                              hipStream_t stream) {
  const float* x       = (const float*)d_in[0];
  const float* h_prev  = (const float*)d_in[1];
  const float* hs_prev = (const float*)d_in[2];
  const float* Wi      = (const float*)d_in[3];
  const float* Wh      = (const float*)d_in[4];
  const float* bi      = (const float*)d_in[5];
  const float* bh      = (const float*)d_in[6];
  const float* W_z     = (const float*)d_in[7];
  const float* U_z     = (const float*)d_in[8];
  const float* Us_z    = (const float*)d_in[9];
  const float* W_r     = (const float*)d_in[10];
  const float* U_r     = (const float*)d_in[11];
  const float* Us_r    = (const float*)d_in[12];
  const float* W_hn    = (const float*)d_in[13];
  const float* U_hn    = (const float*)d_in[14];
  const float* Us_hn   = (const float*)d_in[15];
  unsigned short* wp   = (unsigned short*)d_ws;
  float* out           = (float*)d_out;

  pack_weights<<<(TOT_PACK + 255) / 256, 256, 0, stream>>>(
      Wi, Wh, W_z, W_r, W_hn, U_z, U_r, U_hn, Us_z, Us_r, Us_hn, wp);

  gru_fused<<<BATCH / 64, 128, 0, stream>>>(x, h_prev, hs_prev, bi, bh, wp, out);
}
